// GCN_thnh_47278999995055
// MI455X (gfx1250) — compile-verified, matrix operand order checked by reading, not run on hardware
//
#include <hip/hip_runtime.h>
#include <hip/hip_bf16.h>
#include <stdint.h>

typedef __attribute__((ext_vector_type(16))) _Float16 v16h;
typedef __attribute__((ext_vector_type(8)))  float    v8f;

#define TPB 256

// ---------------------------------------------------------------- utilities
__global__ void fill_f32(float* __restrict__ p, float v, int n) {
    int i = blockIdx.x * blockDim.x + threadIdx.x;
    if (i < n) p[i] = v;
}

// deg[col[e]] += 1  (hardware global_atomic_add_f32; deg pre-filled with 1.0 self-loop)
__global__ void deg_count(const long long* __restrict__ col, float* __restrict__ deg, int E) {
    int e = blockIdx.x * blockDim.x + threadIdx.x;
    if (e < E) {
        long long c = col[e];
        unsafeAtomicAdd(&deg[c], 1.0f);
    }
}

__global__ void deg_to_dinv(float* __restrict__ d, int n) {
    int i = blockIdx.x * blockDim.x + threadIdx.x;
    if (i < n) d[i] = rsqrtf(d[i]);   // deg >= 1 always (self-loop)
}

// ------------------------------------------------- dense transform via WMMA
// H[N,NOUT] = X[N,KIN] @ W[KIN,NOUT], KIN<=32, NOUT<=16 (compile-time).
// One wave per 16-node tile, single v_wmma_f32_16x16x32_f16 (f32 accumulate).
//
// ISA layouts (wave32):
//   A 16x32 f16 : lane l carries row M=l&15; lanes 0-15 hold K 0-7 (elems 0-7)
//                 and K 16-23 (elems 8-15); lanes 16-31 hold K 8-15 / 24-31.
//   B 32x16 f16 : same K striping, lane carries column N=l&15.
//   C/D 16x16 f32: elem r of lane l -> (M = r + 8*(l>>4), N = l&15).
template <int KIN, int NOUT>
__global__ void gemm_wmma(const float* __restrict__ X, const float* __restrict__ W,
                          float* __restrict__ H, int N) {
    int wave   = (blockIdx.x * blockDim.x + threadIdx.x) >> 5;
    int lane   = threadIdx.x & 31;
    int nTiles = (N + 15) >> 4;
    if (wave >= nTiles) return;                 // wave-uniform: EXEC stays all-ones

    const bool hi = (lane & 16) != 0;           // which K half this lane carries
    const int  m  = lane & 15;                  // A row / B,D column index

    // ---- A tile (zero-padded K); compile-time k predicates per lane half
    v16h a = {};
    long long node = (long long)wave * 16 + m;
    if (node < N) {
        const float* xr = X + node * (long long)KIN;
        if (KIN == 4) {                         // aligned: stride 16B
            if (!hi) {
                float4 v = *(const float4*)xr;
                a[0] = (_Float16)v.x; a[1] = (_Float16)v.y;
                a[2] = (_Float16)v.z; a[3] = (_Float16)v.w;
            }
        } else if (!hi) {                       // K = 0..7 and 16..23
            #pragma unroll
            for (int j = 0; j < 8; ++j) { if (j      < KIN) a[j]     = (_Float16)xr[j];      }
            #pragma unroll
            for (int j = 0; j < 8; ++j) { if (16 + j < KIN) a[8 + j] = (_Float16)xr[16 + j]; }
        } else {                                // K = 8..15 and 24..31
            #pragma unroll
            for (int j = 0; j < 8; ++j) { if (8  + j < KIN) a[j]     = (_Float16)xr[8 + j];  }
            #pragma unroll
            for (int j = 0; j < 8; ++j) { if (24 + j < KIN) a[8 + j] = (_Float16)xr[24 + j]; }
        }
    }

    // ---- B tile: W[KIN,NOUT] zero-padded to 32x16
    v16h bm = {};
    if (m < NOUT) {
        if (!hi) {
            #pragma unroll
            for (int j = 0; j < 8; ++j) { if (j      < KIN) bm[j]     = (_Float16)W[j        * NOUT + m]; }
            #pragma unroll
            for (int j = 0; j < 8; ++j) { if (16 + j < KIN) bm[8 + j] = (_Float16)W[(16 + j) * NOUT + m]; }
        } else {
            #pragma unroll
            for (int j = 0; j < 8; ++j) { if (8  + j < KIN) bm[j]     = (_Float16)W[(8 + j)  * NOUT + m]; }
            #pragma unroll
            for (int j = 0; j < 8; ++j) { if (24 + j < KIN) bm[8 + j] = (_Float16)W[(24 + j) * NOUT + m]; }
        }
    }

    // ---- D = A x B + 0
    v8f c = {};
    c = __builtin_amdgcn_wmma_f32_16x16x32_f16(
        /*neg_a=*/false, a, /*neg_b=*/false, bm,
        /*c_mod=*/(short)0, c, /*reuse_a=*/false, /*reuse_b=*/false);

    // ---- store D: wave-uniform full-tile fast path (no per-store exec masking)
    bool fullTile = ((long long)(wave + 1) * 16 <= (long long)N);   // uniform
    if (m < NOUT) {
        long long base = (long long)wave * 16 + (hi ? 8 : 0);
        if (fullTile) {
            #pragma unroll
            for (int r = 0; r < 8; ++r) H[(base + r) * NOUT + m] = c[r];
        } else {
            #pragma unroll
            for (int r = 0; r < 8; ++r) { if (base + r < N) H[(base + r) * NOUT + m] = c[r]; }
        }
    }
}

// ------------------------------------------------- edge-wise message scatter
// agg[col] += h[row] * dinv[row]*dinv[col]   (F floats per node)
// h rows are F*4-byte aligned (workspace layout), so gather as one b128/b64.
template <int F>
__global__ void edge_agg(const long long* __restrict__ row, const long long* __restrict__ col,
                         const float* __restrict__ dinv, const float* __restrict__ h,
                         float* __restrict__ agg, int E) {
    int e = blockIdx.x * blockDim.x + threadIdx.x;
    if (e >= E) return;
    long long r = row[e];
    long long c = col[e];
    float norm = dinv[r] * dinv[c];
    float* ac = agg + c * (long long)F;
    if (F == 4) {
        float4 hv = *(const float4*)(h + r * 4LL);
        unsafeAtomicAdd(&ac[0], hv.x * norm);
        unsafeAtomicAdd(&ac[1], hv.y * norm);
        unsafeAtomicAdd(&ac[2], hv.z * norm);
        unsafeAtomicAdd(&ac[3], hv.w * norm);
    } else {
        float2 hv = *(const float2*)(h + r * 2LL);
        unsafeAtomicAdd(&ac[0], hv.x * norm);
        unsafeAtomicAdd(&ac[1], hv.y * norm);
    }
}

// agg = tanh(agg + dinv^2 * h + b)    (in place; becomes next layer's input)
template <int F>
__global__ void finalize_tanh(float* __restrict__ agg, const float* __restrict__ h,
                              const float* __restrict__ dinv, const float* __restrict__ b,
                              int N) {
    int i = blockIdx.x * blockDim.x + threadIdx.x;
    if (i < N * F) {
        int node = i / F, f = i & (F - 1);      // F is a power of two
        float di = dinv[node];
        agg[i] = tanhf(agg[i] + di * di * h[i] + b[f]);
    }
}

// out = agg + dinv^2 * h + b
template <int F>
__global__ void finalize_out(const float* __restrict__ agg, const float* __restrict__ h,
                             const float* __restrict__ dinv, const float* __restrict__ b,
                             float* __restrict__ out, int N) {
    int i = blockIdx.x * blockDim.x + threadIdx.x;
    if (i < N * F) {
        int node = i / F, f = i & (F - 1);
        float di = dinv[node];
        out[i] = agg[i] + di * di * h[i] + b[f];
    }
}

// --------------------------------------------------------------------------
extern "C" void kernel_launch(void* const* d_in, const int* in_sizes, int n_in,
                              void* d_out, int out_size, void* d_ws, size_t ws_size,
                              hipStream_t stream) {
    const float*     x    = (const float*)d_in[0];
    const long long* ei   = (const long long*)d_in[1];   // int64 [2, E]
    const float*     W1   = (const float*)d_in[2];
    const float*     b1   = (const float*)d_in[3];
    const float*     W2   = (const float*)d_in[4];
    const float*     b2   = (const float*)d_in[5];

    int N = in_sizes[0] / 9;
    int E = in_sizes[1] / 2;
    const long long* rowp = ei;        // source
    const long long* colp = ei + E;    // target

    // workspace layout (floats): [dinv: N][h1: 4N (reused: h2 2N | agg2 2N)][agg1: 4N]
    float* ws   = (float*)d_ws;
    float* dinv = ws;
    float* h1   = ws + (long long)N;
    float* agg1 = ws + 5LL * N;        // after finalize_tanh holds layer-2 input
    float* h2   = h1;                  // h1 dead after finalize_tanh
    float* agg2 = h1 + 2LL * N;

    int gN  = (N + TPB - 1) / TPB;
    int g2N = (2 * N + TPB - 1) / TPB;
    int g4N = (4 * N + TPB - 1) / TPB;
    int gE  = (E + TPB - 1) / TPB;
    int nTiles = (N + 15) / 16;
    int gGemm  = (nTiles * 32 + TPB - 1) / TPB;   // one wave32 per 16-node tile

    // degree / normalization
    fill_f32   <<<gN, TPB, 0, stream>>>(dinv, 1.0f, N);          // self-loop
    deg_count  <<<gE, TPB, 0, stream>>>(colp, dinv, E);
    deg_to_dinv<<<gN, TPB, 0, stream>>>(dinv, N);

    // layer 1: h1 = x @ W1 ; scatter ; tanh(agg + dinv^2*h1 + b1)
    gemm_wmma<9, 4> <<<gGemm, TPB, 0, stream>>>(x, W1, h1, N);
    fill_f32        <<<g4N, TPB, 0, stream>>>(agg1, 0.0f, 4 * N);
    edge_agg<4>     <<<gE, TPB, 0, stream>>>(rowp, colp, dinv, h1, agg1, E);
    finalize_tanh<4><<<g4N, TPB, 0, stream>>>(agg1, h1, dinv, b1, N);

    // layer 2: h2 = h @ W2 ; scatter ; out = agg + dinv^2*h2 + b2
    gemm_wmma<4, 2> <<<gGemm, TPB, 0, stream>>>(agg1, W2, h2, N);
    fill_f32        <<<g2N, TPB, 0, stream>>>(agg2, 0.0f, 2 * N);
    edge_agg<2>     <<<gE, TPB, 0, stream>>>(rowp, colp, dinv, h2, agg2, E);
    finalize_out<2> <<<g2N, TPB, 0, stream>>>(agg2, h2, dinv, b2, (float*)d_out, N);
}
